// Transformer_Block_78932908966373
// MI455X (gfx1250) — matrix operand order repaired and checked by reading, not compiled, not run on hardware
//
#include <hip/hip_runtime.h>
#include <hip/hip_bf16.h>

// ---------------------------------------------------------------------------
// Transformer block for MI455X (gfx1250, wave32, WMMA).
// All GEMM-shaped math runs through v_wmma_f32_16x16x32_bf16 (bf16 in, f32
// accumulate). Attention is flash-style (online softmax, causal), never
// materializing the 2048x2048 score matrix.
// ---------------------------------------------------------------------------

typedef __attribute__((ext_vector_type(16))) __bf16  v16bf;
typedef __attribute__((ext_vector_type(8)))  float   v8f;
typedef __attribute__((ext_vector_type(8)))  unsigned v8u;

#define BN   8
#define TN   2048
#define CN   768
#define CHN  3072
#define MROW (BN * TN)     // 16384 rows

__device__ __forceinline__ unsigned short f2bf(float f) {
    unsigned u = __builtin_bit_cast(unsigned, f);
    u += 0x7FFFu + ((u >> 16) & 1u);          // round-to-nearest-even
    return (unsigned short)(u >> 16);
}
__device__ __forceinline__ unsigned packbf(float lo, float hi) {
    return (unsigned)f2bf(lo) | ((unsigned)f2bf(hi) << 16);
}
// A-fragment K offset, contiguous order (k = 16*half + 2*v), matching B
__device__ __forceinline__ int akk(int v, int half) {
    return half * 16 + 2 * v;
}
__device__ __forceinline__ v8f wmma_bf16(const unsigned a[8], const unsigned b[8], v8f c) {
    v8u av, bv;
#pragma unroll
    for (int i = 0; i < 8; ++i) { av[i] = a[i]; bv[i] = b[i]; }
    return __builtin_amdgcn_wmma_f32_16x16x32_bf16(
        false, __builtin_bit_cast(v16bf, av),
        false, __builtin_bit_cast(v16bf, bv),
        (short)0, c, false, false);
}

// --------------------------------------------------------------------------
// Weight pack: fp32 [K,N] -> uint32 [K/2, N], each dword = {bf16 W[k],W[k+1]}
// so a B-matrix fragment VGPR (pair of consecutive K for one column) is a
// single aligned dword load.
// --------------------------------------------------------------------------
__global__ void packW(const float* __restrict__ W, unsigned* __restrict__ out,
                      int K, int N) {
    int idx = blockIdx.x * 256 + threadIdx.x;
    int total = (K >> 1) * N;
    if (idx >= total) return;
    int r = idx / N, n = idx - r * N;
    out[idx] = packbf(W[(size_t)(2 * r) * N + n], W[(size_t)(2 * r + 1) * N + n]);
}

// --------------------------------------------------------------------------
// LayerNorm: fp32 [rows,768] -> bf16 [rows,768]. One block per row.
// --------------------------------------------------------------------------
__global__ __launch_bounds__(256) void ln_k(const float* __restrict__ x,
                                            const float* __restrict__ g,
                                            const float* __restrict__ bia,
                                            unsigned short* __restrict__ out) {
    __shared__ float r1[256], r2[256];
    int tid = threadIdx.x;
    const float* xr = x + (size_t)blockIdx.x * CN;
    float v0 = xr[tid], v1 = xr[tid + 256], v2 = xr[tid + 512];
    r1[tid] = v0 + v1 + v2;
    r2[tid] = v0 * v0 + v1 * v1 + v2 * v2;
    __syncthreads();
    for (int off = 128; off; off >>= 1) {
        if (tid < off) { r1[tid] += r1[tid + off]; r2[tid] += r2[tid + off]; }
        __syncthreads();
    }
    float mu  = r1[0] * (1.0f / CN);
    float var = r2[0] * (1.0f / CN) - mu * mu;
    float rs  = rsqrtf(var + 1e-5f);
    unsigned short* o = out + (size_t)blockIdx.x * CN;
    o[tid]       = f2bf((v0 - mu) * rs * g[tid]       + bia[tid]);
    o[tid + 256] = f2bf((v1 - mu) * rs * g[tid + 256] + bia[tid + 256]);
    o[tid + 512] = f2bf((v2 - mu) * rs * g[tid + 512] + bia[tid + 512]);
}

// --------------------------------------------------------------------------
// WMMA GEMM: out[M,N] = act(A[M,K] @ W[K,N] + bias) + resid
//   A   : bf16 row-major, read as paired dwords (consecutive K)
//   Bp  : K-paired bf16 weight dwords [K/2, N]
//   mode: 0 = f32 out, 1 = bf16 row-major, 2 = bf16 row-paired dwords
// Block = 8 waves, tile 128(M) x 64(N); each wave a 16x64 strip.
// --------------------------------------------------------------------------
__global__ __launch_bounds__(256) void gemm_bf16(const unsigned* __restrict__ A,
                                                 const unsigned* __restrict__ Bp,
                                                 const float* __restrict__ bias,
                                                 const float* __restrict__ resid,
                                                 void* __restrict__ out,
                                                 int N, int K, int gelu, int mode) {
    int lane = threadIdx.x & 31, w = threadIdx.x >> 5;
    int m16 = lane & 15, half = lane >> 4;
    int mbase = blockIdx.y * 128 + w * 16;
    int nbase = blockIdx.x * 64;

    v8f acc[4];
#pragma unroll
    for (int t = 0; t < 4; ++t)
#pragma unroll
        for (int v = 0; v < 8; ++v) acc[t][v] = 0.f;

    const unsigned* arow = A + (size_t)(mbase + m16) * (K >> 1);
    for (int k0 = 0; k0 < K; k0 += 32) {
        unsigned af[8];
#pragma unroll
        for (int v = 0; v < 8; ++v) af[v] = arow[(k0 + akk(v, half)) >> 1];
#pragma unroll
        for (int t = 0; t < 4; ++t) {
            unsigned bf[8];
#pragma unroll
            for (int v = 0; v < 8; ++v) {
                int kk = k0 + half * 16 + 2 * v;            // 32x16 B layout
                bf[v] = Bp[(size_t)(kk >> 1) * N + nbase + t * 16 + m16];
            }
            acc[t] = wmma_bf16(af, bf, acc[t]);
        }
    }

#pragma unroll
    for (int t = 0; t < 4; ++t) {
        int col = nbase + t * 16 + m16;
        float bv = bias ? bias[col] : 0.f;
        float res[8];
#pragma unroll
        for (int v = 0; v < 8; ++v) {
            float xv = acc[t][v] + bv;
            if (gelu) xv = 0.5f * xv * (1.f + erff(xv * 0.70710678118654752f));
            if (resid) xv += resid[(size_t)(mbase + half * 8 + v) * N + col];
            res[v] = xv;
        }
        if (mode == 0) {
            float* o = (float*)out;
#pragma unroll
            for (int v = 0; v < 8; ++v)
                o[(size_t)(mbase + half * 8 + v) * N + col] = res[v];
        } else if (mode == 1) {
            unsigned short* o = (unsigned short*)out;
#pragma unroll
            for (int v = 0; v < 8; ++v)
                o[(size_t)(mbase + half * 8 + v) * N + col] = f2bf(res[v]);
        } else {
            unsigned* o = (unsigned*)out;   // row-paired (for V operand of P@V)
#pragma unroll
            for (int v = 0; v < 8; v += 2)
                o[(size_t)((mbase + half * 8 + v) >> 1) * N + col] =
                    packbf(res[v], res[v + 1]);
        }
    }
}

// --------------------------------------------------------------------------
// Flash attention (causal, single head, d = 768).
// One block per (b, 16-query tile). 8 waves split d into 96-wide slices:
//   S step (32 keys): each wave WMMAs its c-slice partial, ds_add_f32 reduce,
//   wave0 does masked online softmax (shfl_xor butterflies in 16-lane halves),
//   broadcasts P + rescale via LDS, all waves WMMA P@V into their d-slice.
// --------------------------------------------------------------------------
__global__ __launch_bounds__(256) void attn_k(const unsigned* __restrict__ Q,
                                              const unsigned* __restrict__ Kb,
                                              const unsigned* __restrict__ Vp,
                                              unsigned short* __restrict__ Y) {
    __shared__ float sS[16 * 32];
    __shared__ float sStat[16];
    int lane = threadIdx.x & 31, w = threadIdx.x >> 5;
    int m16 = lane & 15, half = lane >> 4;
    int qt = blockIdx.x, b = blockIdx.y;
    const float scale = 0.036084391824351615f;   // 1/sqrt(768)

    // Q fragments for this wave's 96-wide c-slice (3 k-steps of 32)
    unsigned qf[3][8];
    size_t qrow = ((size_t)b * TN + qt * 16 + m16) * CN;
#pragma unroll
    for (int i = 0; i < 3; ++i)
#pragma unroll
        for (int v = 0; v < 8; ++v)
            qf[i][v] = Q[(qrow + w * 96 + i * 32 + akk(v, half)) >> 1];

    v8f o[6];
#pragma unroll
    for (int s = 0; s < 6; ++s)
#pragma unroll
        for (int v = 0; v < 8; ++v) o[s][v] = 0.f;
    float mrow[8], lrow[8];
#pragma unroll
    for (int v = 0; v < 8; ++v) { mrow[v] = -1e30f; lrow[v] = 0.f; }

    int nsteps = qt / 2 + 1;                 // keys 0 .. 32*nsteps-1 (causal)
    for (int st = 0; st < nsteps; ++st) {
        int key0 = st * 32;
        __syncthreads();
        sS[threadIdx.x] = 0.f; sS[threadIdx.x + 256] = 0.f;
        __syncthreads();

        // partial S over this wave's c-slice
        v8f sp[2];
#pragma unroll
        for (int t = 0; t < 2; ++t) {
#pragma unroll
            for (int v = 0; v < 8; ++v) sp[t][v] = 0.f;
#pragma unroll
            for (int i = 0; i < 3; ++i) {
                unsigned bf[8];
                size_t krow = ((size_t)b * TN + key0 + t * 16 + m16) * CN;
#pragma unroll
                for (int v = 0; v < 8; ++v)
                    bf[v] = Kb[(krow + w * 96 + i * 32 + half * 16 + 2 * v) >> 1];
                sp[t] = wmma_bf16(qf[i], bf, sp[t]);
            }
        }
#pragma unroll
        for (int t = 0; t < 2; ++t)
#pragma unroll
            for (int v = 0; v < 8; ++v)
                atomicAdd(&sS[(v + half * 8) * 32 + t * 16 + m16], sp[t][v]);
        __syncthreads();

        if (w == 0) {   // online softmax on full S (16x32), causal mask
            float s0[8], s1[8];
#pragma unroll
            for (int v = 0; v < 8; ++v) {
                int m = v + half * 8, qg = qt * 16 + m;
                float a = sS[m * 32 + m16], c = sS[m * 32 + 16 + m16];
                s0[v] = (key0 + m16      <= qg) ? a * scale : -1e30f;
                s1[v] = (key0 + 16 + m16 <= qg) ? c * scale : -1e30f;
            }
            float corr[8];
#pragma unroll
            for (int v = 0; v < 8; ++v) {
                float mx = fmaxf(s0[v], s1[v]);
#pragma unroll
                for (int d = 1; d < 16; d <<= 1)
                    mx = fmaxf(mx, __shfl_xor(mx, d, 32));
                float nm = fmaxf(mrow[v], mx);
                float p0 = __expf(s0[v] - nm), p1 = __expf(s1[v] - nm);
                float rs = p0 + p1;
#pragma unroll
                for (int d = 1; d < 16; d <<= 1)
                    rs += __shfl_xor(rs, d, 32);
                corr[v] = __expf(mrow[v] - nm);
                lrow[v] = lrow[v] * corr[v] + rs;
                mrow[v] = nm;
                s0[v] = p0; s1[v] = p1;
            }
#pragma unroll
            for (int v = 0; v < 8; ++v) {   // write P back (broadcast)
                int m = v + half * 8;
                sS[m * 32 + m16]      = s0[v];
                sS[m * 32 + 16 + m16] = s1[v];
            }
            if (m16 == 0)
#pragma unroll
                for (int v = 0; v < 8; ++v) sStat[half * 8 + v] = corr[v];
        }
        __syncthreads();

        // rescale O, then O += P @ V_slice
        float cr[8];
#pragma unroll
        for (int v = 0; v < 8; ++v) cr[v] = sStat[v + half * 8];
#pragma unroll
        for (int s = 0; s < 6; ++s)
#pragma unroll
            for (int v = 0; v < 8; ++v) o[s][v] *= cr[v];

        unsigned pf[8];                      // P as bf16 A-fragment (16x32)
#pragma unroll
        for (int v = 0; v < 8; ++v) {
            int kk = akk(v, half);
            pf[v] = packbf(sS[m16 * 32 + kk], sS[m16 * 32 + kk + 1]);
        }
#pragma unroll
        for (int s = 0; s < 6; ++s) {
            unsigned bf[8];
            int d0 = w * 96 + s * 16;
#pragma unroll
            for (int v = 0; v < 8; ++v) {
                int kk = half * 16 + 2 * v;  // key index within step
                bf[v] = Vp[(size_t)(((size_t)b * TN + key0 + kk) >> 1) * CN + d0 + m16];
            }
            o[s] = wmma_bf16(pf, bf, o[s]);
        }
    }

    __syncthreads();
    if (w == 0 && m16 == 0)
#pragma unroll
        for (int v = 0; v < 8; ++v) sStat[half * 8 + v] = lrow[v];
    __syncthreads();
    float li[8];
#pragma unroll
    for (int v = 0; v < 8; ++v) li[v] = 1.f / sStat[v + half * 8];
#pragma unroll
    for (int s = 0; s < 6; ++s)
#pragma unroll
        for (int v = 0; v < 8; ++v)
            Y[((size_t)b * TN + qt * 16 + v + half * 8) * CN + w * 96 + s * 16 + m16] =
                f2bf(o[s][v] * li[v]);
}

// --------------------------------------------------------------------------
extern "C" void kernel_launch(void* const* d_in, const int* in_sizes, int n_in,
                              void* d_out, int out_size, void* d_ws, size_t ws_size,
                              hipStream_t stream) {
    (void)in_sizes; (void)n_in; (void)out_size; (void)ws_size;
    const float* x   = (const float*)d_in[0];
    const float* Wq  = (const float*)d_in[1];
    const float* Wk  = (const float*)d_in[2];
    const float* Wv  = (const float*)d_in[3];
    const float* Wp  = (const float*)d_in[4];
    const float* bp  = (const float*)d_in[5];
    const float* g1  = (const float*)d_in[6];
    const float* b1  = (const float*)d_in[7];
    const float* g2  = (const float*)d_in[8];
    const float* b2  = (const float*)d_in[9];
    const float* W1  = (const float*)d_in[10];
    const float* bm1 = (const float*)d_in[11];
    const float* W2  = (const float*)d_in[12];
    const float* bm2 = (const float*)d_in[13];

    // workspace layout (bytes); mid aliases xn/q/k/y, h aliases vp
    const size_t SZ = (size_t)MROW * CN * 2;     // 25 165 824
    char* ws = (char*)d_ws;
    unsigned short* xn  = (unsigned short*)(ws + 0 * SZ);
    unsigned short* qb  = (unsigned short*)(ws + 1 * SZ);
    unsigned short* kb  = (unsigned short*)(ws + 2 * SZ);
    unsigned short* yb  = (unsigned short*)(ws + 3 * SZ);
    unsigned short* mid = (unsigned short*)(ws + 0 * SZ);   // [M, 3072] = 4*SZ
    unsigned*       vp  = (unsigned*)      (ws + 4 * SZ);   // row-paired V
    unsigned short* hb  = (unsigned short*)(ws + 4 * SZ);   // after attn
    float*          x2  = (float*)         (ws + 5 * SZ);   // 2*SZ bytes (f32)
    char* wbase = ws + 7 * SZ;
    const size_t WSQ = (size_t)CN * CN * 2;      // packed 768x768
    unsigned* WqP = (unsigned*)(wbase + 0 * WSQ);
    unsigned* WkP = (unsigned*)(wbase + 1 * WSQ);
    unsigned* WvP = (unsigned*)(wbase + 2 * WSQ);
    unsigned* WpP = (unsigned*)(wbase + 3 * WSQ);
    unsigned* W1P = (unsigned*)(wbase + 4 * WSQ);
    unsigned* W2P = (unsigned*)((char*)W1P + (size_t)CN * CHN * 2);

    const int PT  = (CN * CN / 2 + 255) / 256;       // pack threads (square)
    const int PT2 = (CN * CHN / 2 + 255) / 256;
    packW<<<PT,  256, 0, stream>>>(Wq, WqP, CN, CN);
    packW<<<PT,  256, 0, stream>>>(Wk, WkP, CN, CN);
    packW<<<PT,  256, 0, stream>>>(Wv, WvP, CN, CN);
    packW<<<PT,  256, 0, stream>>>(Wp, WpP, CN, CN);
    packW<<<PT2, 256, 0, stream>>>(W1, W1P, CN, CHN);
    packW<<<PT2, 256, 0, stream>>>(W2, W2P, CHN, CN);

    // xn = LN1(x)
    ln_k<<<MROW, 256, 0, stream>>>(x, g1, b1, xn);

    dim3 gN768(CN / 64, MROW / 128), gN3072(CHN / 64, MROW / 128);
    // q, k bf16 row-major; v row-paired for the P@V B-operand
    gemm_bf16<<<gN768, 256, 0, stream>>>((const unsigned*)xn, WqP, nullptr, nullptr, qb, CN, CN, 0, 1);
    gemm_bf16<<<gN768, 256, 0, stream>>>((const unsigned*)xn, WkP, nullptr, nullptr, kb, CN, CN, 0, 1);
    gemm_bf16<<<gN768, 256, 0, stream>>>((const unsigned*)xn, WvP, nullptr, nullptr, vp, CN, CN, 0, 2);

    // y = causal softmax(q k^T / sqrt(d)) v
    attn_k<<<dim3(TN / 16, BN), 256, 0, stream>>>((const unsigned*)qb, (const unsigned*)kb, vp, yb);

    // x2 = x + y @ Wp + bp   (f32)
    gemm_bf16<<<gN768, 256, 0, stream>>>((const unsigned*)yb, WpP, bp, x, x2, CN, CN, 0, 0);

    // h = LN2(x2)
    ln_k<<<MROW, 256, 0, stream>>>(x2, g2, b2, hb);

    // mid = gelu(h @ W1 + bm1)  (bf16)
    gemm_bf16<<<gN3072, 256, 0, stream>>>((const unsigned*)hb, W1P, bm1, nullptr, mid, CHN, CN, 1, 1);

    // out = x2 + mid @ W2 + bm2  (f32)
    gemm_bf16<<<gN768, 256, 0, stream>>>((const unsigned*)mid, W2P, bm2, x2, d_out, CN, CHN, 0, 0);
}
